// GINet_68453188763931
// MI455X (gfx1250) — matrix operand order repaired and checked by reading, not compiled, not run on hardware
//
#include <hip/hip_runtime.h>
#include <hip/hip_bf16.h>

// ---------------------------------------------------------------------------
// GINet forward for MI455X (gfx1250, wave32, WMMA).
// GEMMs: operands pre-padded/converted to f16 (A row-major [Mp,Kp],
// W transposed to n-major [Np,Kp]); WMMA kernel is branch-free with
// double-buffered LDS: one barrier per K-step, global loads for step k+1
// overlap the WMMAs of step k, prefetch issued two steps ahead.
// Block tile 128x64, 8 waves, each wave = 2x2 fragment outer product
// -> 4x v_wmma_f32_16x16x32_f16 per wave per K-step.
// Graph scatter/gather ops use global float atomics (bandwidth bound).
// ---------------------------------------------------------------------------

typedef __attribute__((ext_vector_type(16))) _Float16 v16h;
typedef __attribute__((ext_vector_type(8)))  _Float16 v8h;
typedef __attribute__((ext_vector_type(8)))  float    v8f;

#define N_NODES 100000
#define N_EDGES 200000
#define N_GRAPH 2000
#define DIM     300
#define N_LAYER 5
#define FEAT    512
#define MAXDEG  4
#define EPS_BN  1e-5f

// ======================= padded f16 conversion =============================

__global__ void convA_kernel(const float* __restrict__ in, _Float16* __restrict__ out,
                             int M, int K, int Kp, long long total)
{
    long long idx = (long long)blockIdx.x * blockDim.x + threadIdx.x;
    if (idx >= total) return;
    int r = (int)(idx / Kp);
    int k = (int)(idx % Kp);
    float v = (r < M && k < K) ? in[(size_t)r * K + k] : 0.0f;
    out[idx] = (_Float16)v;
}

__global__ void convW_kernel(const float* __restrict__ w, _Float16* __restrict__ out,
                             int K, int Nout, int Kp, long long total)
{
    long long idx = (long long)blockIdx.x * blockDim.x + threadIdx.x;
    if (idx >= total) return;
    int n = (int)(idx / Kp);
    int k = (int)(idx % Kp);
    float v = (k < K && n < Nout) ? w[(size_t)k * Nout + n] : 0.0f;
    out[idx] = (_Float16)v;
}

// ======================= branch-free WMMA GEMM =============================
// C[M x Nout] = act( Ah[Mp,Kp] @ Wh[Np,Kp]^T + bias (+ addend) )
// block = 256 threads (8 waves). Tile BM=128, BN=64, TK=32.
// Wave w: rows (w&3)*32, cols (w>>2)*32 -> 2 A frags x 2 B frags = 4 WMMAs.
// LDS row stride 40 halfs (80 B = 20 banks): 16 consecutive rows map to
// distinct bank groups; all vector LDS ops are 16B aligned. Double-buffered.

#define TK 32
#define LDSTR 40

__global__ __launch_bounds__(256)
void gemm_wmma_kernel(const _Float16* __restrict__ Ah,
                      const _Float16* __restrict__ Wh,
                      const float* __restrict__ bias,
                      const float* __restrict__ addend,
                      float* __restrict__ C,
                      int M, int Nout, int Kp, int do_relu)
{
    __shared__ __align__(16) _Float16 At[2][128][LDSTR];
    __shared__ __align__(16) _Float16 Bt[2][64][LDSTR];   // Bt[.][n][k]

    const int m0 = blockIdx.y * 128;
    const int n0 = blockIdx.x * 64;
    const int tid  = threadIdx.x;
    const int lane = tid & 31;
    const int wv   = tid >> 5;          // 0..7
    const int mi   = (wv & 3) * 32;     // wave row base within tile
    const int ni   = (wv >> 2) * 32;    // wave col base within tile
    const int half = lane >> 4;         // 0: lanes 0-15, 1: lanes 16-31
    const int l16  = lane & 15;

    // staging coordinates: A is 128 rows x 32 k (2 chunks/thread),
    // B is 64 rows x 32 k (1 chunk/thread); 8 halfs per chunk.
    const int sr = tid >> 2;            // 0..63
    const int sk = (tid & 3) * 8;       // 0,8,16,24

    const _Float16* aPtr0 = Ah + (size_t)(m0 + sr)      * Kp + sk;
    const _Float16* aPtr1 = Ah + (size_t)(m0 + sr + 64) * Kp + sk;
    const _Float16* bPtr  = Wh + (size_t)(n0 + sr)      * Kp + sk;

    v8f acc00 = {}, acc01 = {}, acc10 = {}, acc11 = {};

    const int ksteps = Kp / TK;

    // prologue: fetch tile 0 into registers
    v8h av0 = *(const v8h*)(aPtr0);
    v8h av1 = *(const v8h*)(aPtr1);
    v8h bv  = *(const v8h*)(bPtr);

    for (int ks = 0; ks < ksteps; ++ks) {
        const int cur = ks & 1;
        // store current tile (WAR on buf[cur] protected by barrier of ks-1)
        *(v8h*)&At[cur][sr][sk]      = av0;
        *(v8h*)&At[cur][sr + 64][sk] = av1;
        *(v8h*)&Bt[cur][sr][sk]      = bv;

        // issue next tile's global loads; wait deferred to next ds_store
        if (ks + 1 < ksteps) {
            av0 = *(const v8h*)(aPtr0 + (size_t)(ks + 1) * TK);
            av1 = *(const v8h*)(aPtr1 + (size_t)(ks + 1) * TK);
            bv  = *(const v8h*)(bPtr  + (size_t)(ks + 1) * TK);
            if (ks + 2 < ksteps) {
                __builtin_prefetch(aPtr0 + (size_t)(ks + 2) * TK, 0, 3);
                __builtin_prefetch(aPtr1 + (size_t)(ks + 2) * TK, 0, 3);
                __builtin_prefetch(bPtr  + (size_t)(ks + 2) * TK, 0, 3);
            }
        }
        __syncthreads();

        // ---- A fragments (16-bit A 16x32 layout):
        // half=0: VGPR0..3 -> K 0..7,  VGPR4..7 -> K 16..23
        // half=1: VGPR0..3 -> K 8..15, VGPR4..7 -> K 24..31
        v8h a0lo = *(const v8h*)&At[cur][mi      + l16][half * 8];
        v8h a0hi = *(const v8h*)&At[cur][mi      + l16][16 + half * 8];
        v8h a1lo = *(const v8h*)&At[cur][mi + 16 + l16][half * 8];
        v8h a1hi = *(const v8h*)&At[cur][mi + 16 + l16][16 + half * 8];
        v16h af0, af1;
        #pragma unroll
        for (int i = 0; i < 8; ++i) {
            af0[i] = a0lo[i]; af0[i + 8] = a0hi[i];
            af1[i] = a1lo[i]; af1[i + 8] = a1hi[i];
        }

        // ---- B fragments (same K order as A: half -> K 0..7,16..23 / 8..15,24..31)
        v8h b0lo = *(const v8h*)&Bt[cur][ni      + l16][half * 8];
        v8h b0hi = *(const v8h*)&Bt[cur][ni      + l16][16 + half * 8];
        v8h b1lo = *(const v8h*)&Bt[cur][ni + 16 + l16][half * 8];
        v8h b1hi = *(const v8h*)&Bt[cur][ni + 16 + l16][16 + half * 8];
        v16h bf0, bf1;
        #pragma unroll
        for (int i = 0; i < 8; ++i) {
            bf0[i] = b0lo[i]; bf0[i + 8] = b0hi[i];
            bf1[i] = b1lo[i]; bf1[i + 8] = b1hi[i];
        }

        acc00 = __builtin_amdgcn_wmma_f32_16x16x32_f16(
                    false, af0, false, bf0, (short)0, acc00, false, false);
        acc01 = __builtin_amdgcn_wmma_f32_16x16x32_f16(
                    false, af0, false, bf1, (short)0, acc01, false, false);
        acc10 = __builtin_amdgcn_wmma_f32_16x16x32_f16(
                    false, af1, false, bf0, (short)0, acc10, false, false);
        acc11 = __builtin_amdgcn_wmma_f32_16x16x32_f16(
                    false, af1, false, bf1, (short)0, acc11, false, false);
    }

    // ---- epilogue: C/D layout: VGPR r -> M = r + 8*half, lane -> N = l16
    const bool hasAdd = (addend != nullptr);
    #pragma unroll
    for (int msub = 0; msub < 2; ++msub) {
        #pragma unroll
        for (int nsub = 0; nsub < 2; ++nsub) {
            const int gn = n0 + ni + nsub * 16 + l16;
            if (gn >= Nout) continue;
            const float b = bias[gn];
            const v8f& a = msub ? (nsub ? acc11 : acc10)
                                : (nsub ? acc01 : acc00);
            #pragma unroll
            for (int r = 0; r < 8; ++r) {
                const int gm = m0 + mi + msub * 16 + r + 8 * half;
                if (gm < M) {
                    float v = a[r] + b;
                    if (hasAdd) v += addend[(size_t)gm * Nout + gn];
                    if (do_relu) v = fmaxf(v, 0.0f);
                    C[(size_t)gm * Nout + gn] = v;
                }
            }
        }
    }
}

// ======================= graph / pointwise kernels =========================

__global__ void degree_kernel(const int* __restrict__ dst, int* __restrict__ counts)
{
    int e = blockIdx.x * blockDim.x + threadIdx.x;
    if (e < N_EDGES) atomicAdd(&counts[dst[e]], 1);
}

__global__ void gauss_kernel(const float* __restrict__ dist,
                             const int* __restrict__ dst,
                             const float* __restrict__ g_means,
                             const float* __restrict__ g_stds,
                             float* __restrict__ h_in)
{
    long long idx = (long long)blockIdx.x * blockDim.x + threadIdx.x;
    if (idx >= (long long)N_EDGES * DIM) return;
    int e = (int)(idx / DIM);
    int d = (int)(idx % DIM);
    float sd = fabsf(g_stds[d]) + 0.01f;
    float z  = (dist[e] - g_means[d]) / sd;
    float a  = 2.5066256f; // sqrt(2*3.14159)
    float g  = __expf(-0.5f * z * z) / (a * sd);
    atomicAdd(&h_in[(size_t)dst[e] * DIM + d], g);
}

__global__ void hin_scale_kernel(const int* __restrict__ counts,
                                 const float* __restrict__ scale_param,
                                 float* __restrict__ h_in)
{
    long long idx = (long long)blockIdx.x * blockDim.x + threadIdx.x;
    if (idx >= (long long)N_NODES * DIM) return;
    int n = (int)(idx / DIM);
    int d = (int)(idx % DIM);
    int deg = counts[n] - 1;
    deg = deg < 0 ? 0 : (deg > MAXDEG - 1 ? MAXDEG - 1 : deg);
    h_in[idx] *= __expf(scale_param[deg * DIM + d]);
}

// wave-per-node attention fusion over the 9 atom-feature embeddings
__global__ __launch_bounds__(256)
void embed_fuse_kernel(const int* __restrict__ x,
                       const float* __restrict__ atom_emb,   // [9,119,D]
                       const float* __restrict__ att_vec,
                       float* __restrict__ fused)
{
    const int lane = threadIdx.x & 31;
    const int wid  = threadIdx.x >> 5;
    const int n    = blockIdx.x * 8 + wid;
    if (n >= N_NODES) return;

    float score[9];
    #pragma unroll
    for (int f = 0; f < 9; ++f) {
        int xa = x[n * 9 + f];
        const float* emb = atom_emb + ((size_t)f * 119 + xa) * DIM;
        float s = 0.0f;
        for (int d = lane; d < DIM; d += 32) s += emb[d] * att_vec[d];
        #pragma unroll
        for (int off = 16; off > 0; off >>= 1) s += __shfl_down(s, off, 32);
        score[f] = __shfl(s, 0, 32);
    }
    float mx = score[0];
    #pragma unroll
    for (int f = 1; f < 9; ++f) mx = fmaxf(mx, score[f]);
    float ssum = 0.0f;
    float w[9];
    #pragma unroll
    for (int f = 0; f < 9; ++f) { w[f] = __expf(score[f] - mx); ssum += w[f]; }
    float inv = 1.0f / ssum;
    #pragma unroll
    for (int f = 0; f < 9; ++f) w[f] *= inv;

    for (int d = lane; d < DIM; d += 32) {
        float a = 0.0f;
        #pragma unroll
        for (int f = 0; f < 9; ++f) {
            int xa = x[n * 9 + f];
            a += w[f] * atom_emb[((size_t)f * 119 + xa) * DIM + d];
        }
        fused[(size_t)n * DIM + d] = a;
    }
}

__global__ void msg_edges_kernel(const float* __restrict__ h,
                                 const int* __restrict__ src,
                                 const int* __restrict__ dst,
                                 const int* __restrict__ attr,     // [E,3]
                                 const float* __restrict__ e0,     // [5,D]
                                 const float* __restrict__ e1,
                                 const float* __restrict__ e2,
                                 float* __restrict__ agg)
{
    long long idx = (long long)blockIdx.x * blockDim.x + threadIdx.x;
    if (idx >= (long long)N_EDGES * DIM) return;
    int e = (int)(idx / DIM);
    int d = (int)(idx % DIM);
    int a0 = attr[e * 3 + 0], a1 = attr[e * 3 + 1], a2 = attr[e * 3 + 2];
    float v = h[(size_t)src[e] * DIM + d]
            + e0[a0 * DIM + d] + e1[a1 * DIM + d] + e2[a2 * DIM + d];
    atomicAdd(&agg[(size_t)dst[e] * DIM + d], v);
}

__global__ void msg_self_kernel(const float* __restrict__ h,
                                const float* __restrict__ e0,
                                const float* __restrict__ e1,
                                const float* __restrict__ e2,
                                float* __restrict__ agg)
{
    long long idx = (long long)blockIdx.x * blockDim.x + threadIdx.x;
    if (idx >= (long long)N_NODES * DIM) return;
    int d = (int)(idx % DIM);
    // self-loop attr = [4, 0, 0]
    float v = h[idx] + e0[4 * DIM + d] + e1[d] + e2[d];
    atomicAdd(&agg[idx], v);
}

#define BN_ROWS 128
__global__ __launch_bounds__(256)
void bn_stats_kernel(const float* __restrict__ h,
                     float* __restrict__ bsum, float* __restrict__ bsq)
{
    __shared__ float ls[DIM];
    __shared__ float lq[DIM];
    for (int d = threadIdx.x; d < DIM; d += 256) { ls[d] = 0.0f; lq[d] = 0.0f; }
    __syncthreads();
    const int r0 = blockIdx.x * BN_ROWS;
    for (int idx = threadIdx.x; idx < BN_ROWS * DIM; idx += 256) {
        int r = r0 + idx / DIM;
        int d = idx % DIM;
        if (r < N_NODES) {
            float v = h[(size_t)r * DIM + d];
            atomicAdd(&ls[d], v);
            atomicAdd(&lq[d], v * v);
        }
    }
    __syncthreads();
    for (int d = threadIdx.x; d < DIM; d += 256) {
        atomicAdd(&bsum[d], ls[d]);
        atomicAdd(&bsq[d],  lq[d]);
    }
}

__global__ void bn_apply_kernel(float* __restrict__ h,
                                const float* __restrict__ bsum,
                                const float* __restrict__ bsq,
                                const float* __restrict__ gamma,
                                const float* __restrict__ beta,
                                int do_relu)
{
    long long idx = (long long)blockIdx.x * blockDim.x + threadIdx.x;
    if (idx >= (long long)N_NODES * DIM) return;
    int d = (int)(idx % DIM);
    const float invN = 1.0f / (float)N_NODES;
    float mu  = bsum[d] * invN;
    float var = bsq[d] * invN - mu * mu;
    float v = gamma[d] * (h[idx] - mu) * rsqrtf(var + EPS_BN) + beta[d];
    if (do_relu) v = fmaxf(v, 0.0f);
    h[idx] = v;
}

__global__ void pool_cnt_kernel(const int* __restrict__ batch, float* __restrict__ cnt)
{
    int n = blockIdx.x * blockDim.x + threadIdx.x;
    if (n < N_NODES) atomicAdd(&cnt[batch[n]], 1.0f);
}

__global__ void pool_sum_kernel(const float* __restrict__ h,
                                const int* __restrict__ batch,
                                float* __restrict__ sums)
{
    long long idx = (long long)blockIdx.x * blockDim.x + threadIdx.x;
    if (idx >= (long long)N_NODES * DIM) return;
    int n = (int)(idx / DIM);
    int d = (int)(idx % DIM);
    atomicAdd(&sums[(size_t)batch[n] * DIM + d], h[idx]);
}

__global__ void pool_mean_kernel(const float* __restrict__ sums,
                                 const float* __restrict__ cnt,
                                 float* __restrict__ hg)
{
    int idx = blockIdx.x * blockDim.x + threadIdx.x;
    if (idx >= N_GRAPH * DIM) return;
    int g = idx / DIM;
    hg[idx] = sums[idx] / fmaxf(cnt[g], 1.0f);
}

// =============================== driver ====================================

static inline unsigned blks(long long total, int tpb) {
    return (unsigned)((total + tpb - 1) / tpb);
}

struct GemmCtx {
    _Float16* Ah;
    _Float16* Wh;
    hipStream_t stream;
};

// pad + convert operands, then run the branch-free WMMA GEMM
static inline void gemm(const GemmCtx& ctx,
                        const float* A, const float* Wm, const float* bias,
                        const float* addend, float* C,
                        int M, int K, int Nout, int relu)
{
    const int Kp = (K + 31) & ~31;
    const int Np = (Nout + 63) & ~63;
    const int Mp = (M + 127) & ~127;
    const long long aTot = (long long)Mp * Kp;
    const long long wTot = (long long)Np * Kp;
    convA_kernel<<<blks(aTot, 256), 256, 0, ctx.stream>>>(A, ctx.Ah, M, K, Kp, aTot);
    convW_kernel<<<blks(wTot, 256), 256, 0, ctx.stream>>>(Wm, ctx.Wh, K, Nout, Kp, wTot);
    dim3 grid((Nout + 63) / 64, (M + 127) / 128);
    gemm_wmma_kernel<<<grid, 256, 0, ctx.stream>>>(ctx.Ah, ctx.Wh, bias, addend, C,
                                                   M, Nout, Kp, relu);
}

extern "C" void kernel_launch(void* const* d_in, const int* in_sizes, int n_in,
                              void* d_out, int out_size, void* d_ws, size_t ws_size,
                              hipStream_t stream)
{
    (void)in_sizes; (void)n_in; (void)out_size; (void)ws_size;

    const int*   x        = (const int*)  d_in[0];
    const int*   eidx     = (const int*)  d_in[1];   // [2,E]
    const int*   eattr    = (const int*)  d_in[2];   // [E,3]
    const float* edist    = (const float*)d_in[3];
    const int*   batch    = (const int*)  d_in[4];
    const float* atom_emb = (const float*)d_in[5];
    const float* att_vec  = (const float*)d_in[6];
    const float* outlin_w = (const float*)d_in[7];
    const float* outlin_b = (const float*)d_in[8];
    const float* g_means  = (const float*)d_in[9];
    const float* g_stds   = (const float*)d_in[10];
    const float* scale_p  = (const float*)d_in[11];
    const float* edge_emb = (const float*)d_in[12];  // [L,3,5,D]
    const float* mlp_w1   = (const float*)d_in[13];  // [L,D,2D]
    const float* mlp_b1   = (const float*)d_in[14];  // [L,2D]
    const float* mlp_w2   = (const float*)d_in[15];  // [L,2D,D]
    const float* mlp_b2   = (const float*)d_in[16];  // [L,D]
    const float* bn_gamma = (const float*)d_in[17];
    const float* bn_beta  = (const float*)d_in[18];
    const float* feat_w   = (const float*)d_in[19];  // [D,FEAT]
    const float* feat_b   = (const float*)d_in[20];
    const float* ol_w1    = (const float*)d_in[21];  // [FEAT,FEAT]
    const float* ol_b1    = (const float*)d_in[22];
    const float* ol_w2    = (const float*)d_in[23];  // [FEAT,FEAT/2]
    const float* ol_b2    = (const float*)d_in[24];

    const int* src = eidx;
    const int* dst = eidx + N_EDGES;

    float* out = (float*)d_out;
    float* h_feat = out;                          // [G, FEAT]
    float* out2   = out + (size_t)N_GRAPH * FEAT; // [G, FEAT/2]

    // ---- workspace carve-out (bytes, 256B aligned chunks) ----
    char* base = (char*)d_ws;
    size_t o = 0;
    auto carve = [&](size_t bytes) {
        char* p = base + o;
        o += (bytes + 255) & ~(size_t)255;
        return p;
    };
    // Ah: max Mp*Kp = 100096 * 608 halfs
    _Float16* Ah = (_Float16*)carve((size_t)100096 * 608 * sizeof(_Float16));
    // Wh: max Np*Kp = 512 * 512 halfs
    _Float16* Wh = (_Float16*)carve((size_t)512 * 512 * sizeof(_Float16));
    float* h     = (float*)carve((size_t)N_NODES * DIM * sizeof(float));
    float* agg   = (float*)carve((size_t)N_NODES * DIM * sizeof(float)); // h_in, then agg
    float* fused = (float*)carve((size_t)N_NODES * DIM * sizeof(float));
    float* t     = (float*)carve((size_t)N_NODES * 2 * DIM * sizeof(float));
    float* t2    = (float*)carve((size_t)N_GRAPH * FEAT * sizeof(float));
    float* psum  = (float*)carve((size_t)N_GRAPH * DIM * sizeof(float));
    float* hg    = (float*)carve((size_t)N_GRAPH * DIM * sizeof(float));
    float* pcnt  = (float*)carve((size_t)N_GRAPH * sizeof(float));
    float* bsum  = (float*)carve((size_t)DIM * sizeof(float));
    float* bsq   = (float*)carve((size_t)DIM * sizeof(float));
    int*   counts = (int*)carve((size_t)N_NODES * sizeof(int));

    GemmCtx ctx{Ah, Wh, stream};

    const long long ND = (long long)N_NODES * DIM;
    const long long ED = (long long)N_EDGES * DIM;

    // ---- degree + gaussian RBF into h_in (agg buffer) ----
    hipMemsetAsync(counts, 0, (size_t)N_NODES * sizeof(int), stream);
    hipMemsetAsync(agg, 0, (size_t)ND * sizeof(float), stream);
    degree_kernel<<<blks(N_EDGES, 256), 256, 0, stream>>>(dst, counts);
    gauss_kernel<<<blks(ED, 256), 256, 0, stream>>>(edist, dst, g_means, g_stds, agg);
    hin_scale_kernel<<<blks(ND, 256), 256, 0, stream>>>(counts, scale_p, agg);

    // ---- attention fusion + first GEMM (h = fused@W + b + h_in) ----
    embed_fuse_kernel<<<blks(N_NODES, 8), 256, 0, stream>>>(x, atom_emb, att_vec, fused);
    gemm(ctx, fused, outlin_w, outlin_b, agg, h, N_NODES, DIM, DIM, 0);

    // ---- 5 message-passing layers ----
    for (int l = 0; l < N_LAYER; ++l) {
        const float* e0 = edge_emb + ((size_t)(l * 3 + 0) * 5) * DIM;
        const float* e1 = edge_emb + ((size_t)(l * 3 + 1) * 5) * DIM;
        const float* e2 = edge_emb + ((size_t)(l * 3 + 2) * 5) * DIM;

        hipMemsetAsync(agg, 0, (size_t)ND * sizeof(float), stream);
        msg_edges_kernel<<<blks(ED, 256), 256, 0, stream>>>(h, src, dst, eattr,
                                                            e0, e1, e2, agg);
        msg_self_kernel<<<blks(ND, 256), 256, 0, stream>>>(h, e0, e1, e2, agg);

        gemm(ctx, agg, mlp_w1 + (size_t)l * DIM * 2 * DIM, mlp_b1 + (size_t)l * 2 * DIM,
             nullptr, t, N_NODES, DIM, 2 * DIM, 1);
        gemm(ctx, t, mlp_w2 + (size_t)l * 2 * DIM * DIM, mlp_b2 + (size_t)l * DIM,
             nullptr, h, N_NODES, 2 * DIM, DIM, 0);

        hipMemsetAsync(bsum, 0, DIM * sizeof(float), stream);
        hipMemsetAsync(bsq,  0, DIM * sizeof(float), stream);
        bn_stats_kernel<<<blks(N_NODES, BN_ROWS), 256, 0, stream>>>(h, bsum, bsq);
        bn_apply_kernel<<<blks(ND, 256), 256, 0, stream>>>(
            h, bsum, bsq, bn_gamma + (size_t)l * DIM, bn_beta + (size_t)l * DIM,
            (l < N_LAYER - 1) ? 1 : 0);
    }

    // ---- mean pooling per graph ----
    hipMemsetAsync(psum, 0, (size_t)N_GRAPH * DIM * sizeof(float), stream);
    hipMemsetAsync(pcnt, 0, (size_t)N_GRAPH * sizeof(float), stream);
    pool_cnt_kernel<<<blks(N_NODES, 256), 256, 0, stream>>>(batch, pcnt);
    pool_sum_kernel<<<blks(ND, 256), 256, 0, stream>>>(h, batch, psum);
    pool_mean_kernel<<<blks((long long)N_GRAPH * DIM, 256), 256, 0, stream>>>(psum, pcnt, hg);

    // ---- heads: h_feat (output 0) and out (output 1) ----
    gemm(ctx, hg, feat_w, feat_b, nullptr, h_feat, N_GRAPH, DIM, FEAT, 0);
    gemm(ctx, h_feat, ol_w1, ol_b1, nullptr, t2, N_GRAPH, FEAT, FEAT, 1);
    gemm(ctx, t2, ol_w2, ol_b2, nullptr, out2, N_GRAPH, FEAT, FEAT / 2, 0);
}
